// GCNII_23596550324877
// MI455X (gfx1250) — compile-verified, matrix operand order checked by reading, not run on hardware
//
#include <hip/hip_runtime.h>

#define HIDF 128
#define OUTF 64
#define NLAYERS 8
#define ALPHA_C 0.1f
#define BN_EPS_C 1e-5f

typedef __attribute__((ext_vector_type(16))) __bf16 bf16x16;
typedef __attribute__((ext_vector_type(8)))  float  f32x8;

union FragBF {
    bf16x16 v;
    uint4   q[2];
};

__device__ __forceinline__ unsigned short f2bf(float f) {
    unsigned int u = __float_as_uint(f);
    u += 0x7FFFu + ((u >> 16) & 1u);   // round-to-nearest-even
    return (unsigned short)(u >> 16);
}

__device__ __forceinline__ void atomAddF(float* p, float v) {
    unsafeAtomicAdd(p, v);             // global_atomic_add_f32
}

// ---------------- weight conversion ----------------
__global__ void cvt_direct_kernel(const float* __restrict__ src,
                                  unsigned short* __restrict__ dst, int n) {
    int i = blockIdx.x * blockDim.x + threadIdx.x;
    if (i < n) dst[i] = f2bf(src[i]);
}

// conv_w is (L, K, J) applied as h @ W ; store out-major: dst[l][j*K + k] = W[l][k][j]
__global__ void cvt_conv_kernel(const float* __restrict__ conv,
                                unsigned short* __restrict__ dst, int total) {
    int i = blockIdx.x * blockDim.x + threadIdx.x;
    if (i >= total) return;
    int l   = i / (HIDF * HIDF);
    int rem = i - l * (HIDF * HIDF);
    int j   = rem / HIDF;
    int k   = rem - j * HIDF;
    dst[i] = f2bf(conv[(size_t)l * HIDF * HIDF + (size_t)k * HIDF + j]);
}

// ---------------- degree / norm ----------------
__global__ void deg_init_kernel(float* deg, int n) {
    int i = blockIdx.x * blockDim.x + threadIdx.x;
    if (i < n) deg[i] = 1.0f;          // self loop
}
__global__ void deg_count_kernel(const int* __restrict__ dst, float* deg, int e) {
    int i = blockIdx.x * blockDim.x + threadIdx.x;
    if (i < e) atomAddF(&deg[dst[i]], 1.0f);
}
__global__ void deg_rsqrt_kernel(float* deg, int n) {
    int i = blockIdx.x * blockDim.x + threadIdx.x;
    if (i < n) deg[i] = rsqrtf(deg[i]);
}

// ---------------- zero agg + stats ----------------
__global__ void zero_agg_stats_kernel(float4* __restrict__ agg, int n4,
                                      float4* __restrict__ stats /*64 float4*/) {
    int i = blockIdx.x * blockDim.x + threadIdx.x;
    if (i < 64) stats[i] = make_float4(0.f, 0.f, 0.f, 0.f);
    if (i < n4) agg[i] = make_float4(0.f, 0.f, 0.f, 0.f);
}

// ---------------- SpMM: agg[dst] += dinv[src]*dinv[dst]*h[src] ----------------
__global__ void spmm_kernel(const int* __restrict__ src, const int* __restrict__ dst,
                            const float* __restrict__ dinv, const float* __restrict__ h,
                            float* __restrict__ agg, int nE) {
    int gid = blockIdx.x * blockDim.x + threadIdx.x;
    int e = gid >> 5;
    if (e >= nE) return;
    int f = (gid & 31) << 2;
    int s = src[e], d = dst[e];
    float w = dinv[s] * dinv[d];
    const float4 hv = *(const float4*)(h + (size_t)s * HIDF + f);
    float* a = agg + (size_t)d * HIDF + f;
    atomAddF(a + 0, w * hv.x);
    atomAddF(a + 1, w * hv.y);
    atomAddF(a + 2, w * hv.z);
    atomAddF(a + 3, w * hv.w);
}

// ---------------- combine: zc = (1-a)*(agg + dinv^2*h) + a*h0  (in place into agg) ----------------
__global__ void combine_kernel(float4* __restrict__ agg, const float4* __restrict__ h,
                               const float4* __restrict__ h0, const float* __restrict__ dinv,
                               int n4) {
    int i = blockIdx.x * blockDim.x + threadIdx.x;
    if (i >= n4) return;
    int node = i >> 5;                 // 32 float4 per node
    float di = dinv[node];
    float sl = di * di;
    float4 a = agg[i], hv = h[i], hz = h0[i];
    float4 z;
    z.x = (1.f - ALPHA_C) * (a.x + sl * hv.x) + ALPHA_C * hz.x;
    z.y = (1.f - ALPHA_C) * (a.y + sl * hv.y) + ALPHA_C * hz.y;
    z.z = (1.f - ALPHA_C) * (a.z + sl * hv.z) + ALPHA_C * hz.z;
    z.w = (1.f - ALPHA_C) * (a.w + sl * hv.w) + ALPHA_C * hz.w;
    agg[i] = z;
}

// ---------------- WMMA GEMM ----------------
// A: [M x 128] fp32.  BT: [NOUT x 128] bf16 (out-major, so each lane reads
// contiguous 16B chunks matching the 16-bit WMMA K layout).
// MODE 0: out = relu(A@B + bias), out2 = out           (input projection)
// MODE 1: out = mixA*A + mixB*(A@B), accumulate sum/sq (conv layer)
// MODE 2: out = A@B + bias                             (output projection)
template<int NOUT, int MODE>
__global__ __launch_bounds__(NOUT * 2)
void wmma_gemm_kernel(const float* __restrict__ A, const unsigned short* __restrict__ BT,
                      const float* __restrict__ bias, float* __restrict__ out,
                      float* __restrict__ out2, float* __restrict__ gsum,
                      float* __restrict__ gsq, float mixA, float mixB) {
    __shared__ __align__(16) unsigned short A_lds[16 * HIDF];
    __shared__ float s_sum[NOUT];
    __shared__ float s_sq[NOUT];

    const int tid = threadIdx.x;
    const int Mbase = blockIdx.x * 16;

    // stage 16x128 fp32 tile -> bf16 LDS
    const float4* Ab = (const float4*)(A + (size_t)Mbase * HIDF);
    for (int t = tid; t < 16 * HIDF / 4; t += blockDim.x) {
        float4 f = Ab[t];
        unsigned short* p = &A_lds[t * 4];
        p[0] = f2bf(f.x); p[1] = f2bf(f.y); p[2] = f2bf(f.z); p[3] = f2bf(f.w);
    }
    if (MODE == 1) {
        for (int t = tid; t < NOUT; t += blockDim.x) { s_sum[t] = 0.f; s_sq[t] = 0.f; }
    }
    __syncthreads();

    const int lane = tid & 31;
    const int wave = tid >> 5;
    const int half = lane >> 4;
    const int rowL = lane & 15;
    const int col  = wave * 16 + rowL;

    f32x8 c = {0.f, 0.f, 0.f, 0.f, 0.f, 0.f, 0.f, 0.f};
#pragma unroll
    for (int kk = 0; kk < HIDF / 32; ++kk) {
        const int kb = kk * 32 + half * 8;
        FragBF a, b;
        a.q[0] = *(const uint4*)&A_lds[rowL * HIDF + kb];
        a.q[1] = *(const uint4*)&A_lds[rowL * HIDF + kb + 16];
        const unsigned short* bp = BT + (size_t)col * HIDF + kb;
        b.q[0] = *(const uint4*)(bp);
        b.q[1] = *(const uint4*)(bp + 16);
        c = __builtin_amdgcn_wmma_f32_16x16x32_bf16(false, a.v, false, b.v,
                                                    (short)0, c, false, false);
    }

    float lsum = 0.f, lsq = 0.f;
#pragma unroll
    for (int r = 0; r < 8; ++r) {
        const int row = Mbase + half * 8 + r;
        float v = c[r];
        if (MODE == 0) {
            v += bias[col];
            v = fmaxf(v, 0.f);
            out[(size_t)row * NOUT + col]  = v;
            out2[(size_t)row * NOUT + col] = v;
        } else if (MODE == 2) {
            v += bias[col];
            out[(size_t)row * NOUT + col] = v;
        } else {
            const float zc = A[(size_t)row * HIDF + col];
            v = mixA * zc + mixB * v;
            out[(size_t)row * NOUT + col] = v;
            lsum += v; lsq += v * v;
        }
    }
    if (MODE == 1) {
        atomicAdd(&s_sum[col], lsum);   // ds_add_f32
        atomicAdd(&s_sq[col], lsq);
        __syncthreads();
        if (tid < NOUT) {
            atomAddF(&gsum[tid], s_sum[tid]);
            atomAddF(&gsq[tid],  s_sq[tid]);
        }
    }
}

// ---------------- BN finalize: fold gamma/beta into scale/shift ----------------
__global__ void bn_finalize_kernel(const float* __restrict__ gsum, const float* __restrict__ gsq,
                                   const float* __restrict__ gamma, const float* __restrict__ beta,
                                   float* __restrict__ scale, float* __restrict__ shift,
                                   float invN) {
    int f = threadIdx.x;               // 128 threads
    float mean = gsum[f] * invN;
    float var  = gsq[f] * invN - mean * mean;
    float istd = rsqrtf(var + BN_EPS_C);
    float sc = istd * gamma[f];
    scale[f] = sc;
    shift[f] = beta[f] - mean * sc;
}

__global__ void bn_relu_kernel(float4* __restrict__ z, const float* __restrict__ scale,
                               const float* __restrict__ shift, int n4) {
    int i = blockIdx.x * blockDim.x + threadIdx.x;
    if (i >= n4) return;
    int f = (i & 31) << 2;
    float4 v = z[i];
    v.x = fmaxf(v.x * scale[f + 0] + shift[f + 0], 0.f);
    v.y = fmaxf(v.y * scale[f + 1] + shift[f + 1], 0.f);
    v.z = fmaxf(v.z * scale[f + 2] + shift[f + 2], 0.f);
    v.w = fmaxf(v.w * scale[f + 3] + shift[f + 3], 0.f);
    z[i] = v;
}

// ---------------- host launch ----------------
extern "C" void kernel_launch(void* const* d_in, const int* in_sizes, int n_in,
                              void* d_out, int out_size, void* d_ws, size_t ws_size,
                              hipStream_t stream) {
    const float* x      = (const float*)d_in[0];
    const int*   ei     = (const int*)d_in[1];
    const float* lin0_w = (const float*)d_in[2];
    const float* lin0_b = (const float*)d_in[3];
    const float* lin1_w = (const float*)d_in[4];
    const float* lin1_b = (const float*)d_in[5];
    const float* conv_w = (const float*)d_in[6];
    const float* bn_g   = (const float*)d_in[7];
    const float* bn_b   = (const float*)d_in[8];

    const int N = in_sizes[0] / HIDF;      // 50000
    const int E = in_sizes[1] / 2;         // 800000
    const int* e_src = ei;
    const int* e_dst = ei + E;

    // workspace carve-out (256B aligned)
    char* wsp = (char*)d_ws;
    auto carve = [&](size_t bytes) -> void* {
        void* p = (void*)wsp;
        wsp += (bytes + 255) & ~(size_t)255;
        return p;
    };
    float* dinv = (float*)carve((size_t)N * 4);
    float* h0   = (float*)carve((size_t)N * HIDF * 4);
    float* h    = (float*)carve((size_t)N * HIDF * 4);
    float* agg  = (float*)carve((size_t)N * HIDF * 4);
    unsigned short* Wb0 = (unsigned short*)carve((size_t)HIDF * HIDF * 2);
    unsigned short* Wb1 = (unsigned short*)carve((size_t)OUTF * HIDF * 2);
    unsigned short* Wbc = (unsigned short*)carve((size_t)NLAYERS * HIDF * HIDF * 2);
    float* gsum  = (float*)carve(256 * 4);     // sum[128] + sq[128]
    float* gsq   = gsum + HIDF;
    float* scale = (float*)carve(256 * 4);     // scale[128] + shift[128]
    float* shift = scale + HIDF;

    const int TB = 256;
    const int n4     = N * HIDF / 4;
    const int Mtiles = (N + 15) / 16;          // 3125

    // weight prep
    cvt_direct_kernel<<<(HIDF * HIDF + TB - 1) / TB, TB, 0, stream>>>(lin0_w, Wb0, HIDF * HIDF);
    cvt_direct_kernel<<<(OUTF * HIDF + TB - 1) / TB, TB, 0, stream>>>(lin1_w, Wb1, OUTF * HIDF);
    cvt_conv_kernel<<<(NLAYERS * HIDF * HIDF + TB - 1) / TB, TB, 0, stream>>>(
        conv_w, Wbc, NLAYERS * HIDF * HIDF);

    // symmetric normalization coefficients
    deg_init_kernel<<<(N + TB - 1) / TB, TB, 0, stream>>>(dinv, N);
    deg_count_kernel<<<(E + TB - 1) / TB, TB, 0, stream>>>(e_dst, dinv, E);
    deg_rsqrt_kernel<<<(N + TB - 1) / TB, TB, 0, stream>>>(dinv, N);

    // input projection: h = relu(x @ W0^T + b0); h0 = h
    wmma_gemm_kernel<HIDF, 0><<<Mtiles, HIDF * 2, 0, stream>>>(
        x, Wb0, lin0_b, h, h0, nullptr, nullptr, 0.f, 0.f);

    for (int i = 0; i < NLAYERS; ++i) {
        const float beta = logf(0.5f / (float)(i + 1) + 1.0f);

        zero_agg_stats_kernel<<<(n4 + TB - 1) / TB, TB, 0, stream>>>(
            (float4*)agg, n4, (float4*)gsum);

        spmm_kernel<<<((E * 32) + TB - 1) / TB, TB, 0, stream>>>(
            e_src, e_dst, dinv, h, agg, E);

        combine_kernel<<<(n4 + TB - 1) / TB, TB, 0, stream>>>(
            (float4*)agg, (const float4*)h, (const float4*)h0, dinv, n4);

        // z = (1-beta)*zc + beta*(zc @ Wc[i]); write z into h; accumulate BN stats
        wmma_gemm_kernel<HIDF, 1><<<Mtiles, HIDF * 2, 0, stream>>>(
            agg, Wbc + (size_t)i * HIDF * HIDF, nullptr, h, nullptr,
            gsum, gsq, 1.f - beta, beta);

        bn_finalize_kernel<<<1, HIDF, 0, stream>>>(
            gsum, gsq, bn_g + (size_t)i * HIDF, bn_b + (size_t)i * HIDF,
            scale, shift, 1.0f / (float)N);

        bn_relu_kernel<<<(n4 + TB - 1) / TB, TB, 0, stream>>>(
            (float4*)h, scale, shift, n4);
    }

    // output projection
    wmma_gemm_kernel<OUTF, 2><<<Mtiles, OUTF * 2, 0, stream>>>(
        h, Wb1, lin1_b, (float*)d_out, nullptr, nullptr, nullptr, 0.f, 0.f);
}
